// PureWaveAttention_2654289789225
// MI455X (gfx1250) — matrix operand order repaired and checked by reading, not compiled, not run on hardware
//
#include <hip/hip_runtime.h>
#include <hip/hip_bf16.h>
#include <math.h>

// Problem constants (PureWaveAttention reference)
#define Bc  2
#define Tc  2048
#define Dc  1024
#define Hc  16
#define NWc 64
#define HDc 64

typedef __bf16 bf16_t;
typedef __attribute__((ext_vector_type(16))) __bf16 v16bf;
typedef __attribute__((ext_vector_type(8)))  float  v8f;

// ---------------------------------------------------------------------------
// WMMA helpers (CDNA5 16x16x32 bf16, f32 accumulate)
// ---------------------------------------------------------------------------
__device__ __forceinline__ v8f wmma_bf16(v16bf a, v16bf b, v8f c) {
  // (neg_a, A, neg_b, B, c_mod, C, reuse_a, reuse_b)
  return __builtin_amdgcn_wmma_f32_16x16x32_bf16(false, a, false, b, (short)0, c,
                                                 false, false);
}

__device__ __forceinline__ v8f v8f_zero() {
  v8f z = {0.f, 0.f, 0.f, 0.f, 0.f, 0.f, 0.f, 0.f};
  return z;
}

// A fragment (16x32 bf16): lanes 0-15 hold row M=lane with K=0..15 in v[0..15];
// lanes 16-31 hold the same rows with K=16..31 (contiguous, matching B).
// `rowk` = &A[row=lane&15][k_window_base].
__device__ __forceinline__ v16bf afrag_bf16(const bf16_t* rowk, int lane) {
  const int hi = (lane >> 4) & 1;
  v16bf r;
#pragma unroll
  for (int i = 0; i < 8; ++i) {
    r[i]     = rowk[hi * 16 + i];
    r[i + 8] = rowk[hi * 16 + 8 + i];
  }
  return r;
}

// Same A layout but converting from f32 source (global or LDS).
__device__ __forceinline__ v16bf afrag_f32(const float* rowk, int lane) {
  const int hi = (lane >> 4) & 1;
  v16bf r;
#pragma unroll
  for (int i = 0; i < 8; ++i) {
    r[i]     = (bf16_t)rowk[hi * 16 + i];
    r[i + 8] = (bf16_t)rowk[hi * 16 + 8 + i];
  }
  return r;
}

// B fragment (32x16 bf16): lane n (=lane&15) holds column N=n; lanes 0-15
// K=0..15, lanes 16-31 K=16..31, contiguous -> one 32B load.
__device__ __forceinline__ v16bf bfrag_bf16(const bf16_t* p) {
  return *(const v16bf*)p;
}

// C fragment row for element i: lanes 0-15 -> rows 0..7, lanes 16-31 -> 8..15.
__device__ __forceinline__ int crow(int lane, int i) {
  return ((lane >> 4) & 1) * 8 + i;
}

// ---------------------------------------------------------------------------
// Kernel 0: weight convert + transpose  W[k][n] f32 -> WT[n][k] bf16
// ---------------------------------------------------------------------------
__global__ void k_wcvt(const float* __restrict__ W, bf16_t* __restrict__ WT) {
  int idx = blockIdx.x * blockDim.x + threadIdx.x;   // idx = n*K + k
  if (idx >= Dc * Dc) return;
  int n = idx / Dc;
  int k = idx - n * Dc;
  WT[idx] = (bf16_t)W[(long)k * Dc + n];
}

// ---------------------------------------------------------------------------
// Kernel 1: fused wave projection.  Per wave: 16 rows (t) x 64 cols (w) of one
// head; dual GEMM (f and p share the A fragment), epilogue sin + L2 row-norm.
// out layout: qn[b][h][t][w] bf16.
// ---------------------------------------------------------------------------
__global__ void k_waveproj(const float* __restrict__ x,
                           const bf16_t* __restrict__ WfT,
                           const bf16_t* __restrict__ WpT,
                           const float* __restrict__ biasf,
                           const float* __restrict__ biasp,
                           bf16_t* __restrict__ outn) {
  const int lane = threadIdx.x & 31;
  const int wv   = threadIdx.x >> 5;
  const int h    = blockIdx.y;
  const int m    = lane & 15;
  const int hi   = (lane >> 4) & 1;

  const long r0 = (long)blockIdx.x * 64 + (long)wv * 16;  // row in [0, B*T)
  const long r  = r0 + m;                                 // this lane's A row

  v8f cf[4], cp[4];
#pragma unroll
  for (int nt = 0; nt < 4; ++nt) { cf[nt] = v8f_zero(); cp[nt] = v8f_zero(); }

  for (int k0 = 0; k0 < Dc; k0 += 32) {
    v16bf a = afrag_f32(x + r * Dc + k0, lane);
#pragma unroll
    for (int nt = 0; nt < 4; ++nt) {
      const int n = h * NWc + nt * 16 + m;
      cf[nt] = wmma_bf16(a, bfrag_bf16(WfT + (long)n * Dc + k0 + hi * 16), cf[nt]);
      cp[nt] = wmma_bf16(a, bfrag_bf16(WpT + (long)n * Dc + k0 + hi * 16), cp[nt]);
    }
  }

  // Epilogue: w = sin(f*t + p); L2-normalize along the 64-wide wave dim.
  const int bi    = (int)(r0 / Tc);
  const int tbase = (int)(r0 % Tc);

  float wval[4][8];
  float rowsum[8];
#pragma unroll
  for (int i = 0; i < 8; ++i) rowsum[i] = 0.f;

#pragma unroll
  for (int nt = 0; nt < 4; ++nt) {
    const int n   = h * NWc + nt * 16 + m;
    const float bf_ = biasf[n];
    const float bp_ = biasp[n];
#pragma unroll
    for (int i = 0; i < 8; ++i) {
      const int t = tbase + crow(lane, i);
      const float f = cf[nt][i] + bf_;
      const float p = cp[nt][i] + bp_;
      const float w = sinf(f * (float)t + p);
      wval[nt][i] = w;
      rowsum[i] += w * w;
    }
  }
  // Reduce across the 16 lanes of each half (masks<16 stay within the half).
#pragma unroll
  for (int mask = 1; mask < 16; mask <<= 1) {
#pragma unroll
    for (int i = 0; i < 8; ++i)
      rowsum[i] += __shfl_xor(rowsum[i], mask, 32);
  }
  float inv[8];
#pragma unroll
  for (int i = 0; i < 8; ++i)
    inv[i] = 1.f / fmaxf(sqrtf(rowsum[i]), 1e-12f);

  bf16_t* dst = outn + (((long)bi * Hc + h) * Tc) * NWc;
#pragma unroll
  for (int nt = 0; nt < 4; ++nt) {
    const int n = nt * 16 + m;
#pragma unroll
    for (int i = 0; i < 8; ++i) {
      const int t = tbase + crow(lane, i);
      dst[(long)t * NWc + n] = (bf16_t)(wval[nt][i] * inv[i]);
    }
  }
}

// ---------------------------------------------------------------------------
// Kernel 2: v projection, written TRANSPOSED: vT[b][h][d][t] bf16, so the
// attention P@V B-fragment is a contiguous 32B load along t.
// ---------------------------------------------------------------------------
__global__ void k_vproj(const float* __restrict__ x,
                        const bf16_t* __restrict__ WvT,
                        const float* __restrict__ bv,
                        bf16_t* __restrict__ vT) {
  const int lane = threadIdx.x & 31;
  const int wv   = threadIdx.x >> 5;
  const int h    = blockIdx.y;
  const int m    = lane & 15;
  const int hi   = (lane >> 4) & 1;

  const long r0 = (long)blockIdx.x * 64 + (long)wv * 16;
  const long r  = r0 + m;

  v8f c[4];
#pragma unroll
  for (int nt = 0; nt < 4; ++nt) c[nt] = v8f_zero();

  for (int k0 = 0; k0 < Dc; k0 += 32) {
    v16bf a = afrag_f32(x + r * Dc + k0, lane);
#pragma unroll
    for (int nt = 0; nt < 4; ++nt) {
      const int n = h * HDc + nt * 16 + m;
      c[nt] = wmma_bf16(a, bfrag_bf16(WvT + (long)n * Dc + k0 + hi * 16), c[nt]);
    }
  }

  const int bi    = (int)(r0 / Tc);
  const int tbase = (int)(r0 % Tc);
  bf16_t* dst = vT + ((long)bi * Hc + h) * HDc * Tc;
#pragma unroll
  for (int nt = 0; nt < 4; ++nt) {
    const int d = nt * 16 + m;
    const float bias = bv[h * HDc + d];
#pragma unroll
    for (int i = 0; i < 8; ++i) {
      const int t = tbase + crow(lane, i);
      dst[(long)d * Tc + t] = (bf16_t)(c[nt][i] + bias);
    }
  }
}

// ---------------------------------------------------------------------------
// Kernel 3: causal linear attention. One wave owns a 16-row q tile, streams s
// in steps of 32: S = QnKn^T (4 WMMAs), causal mask, LDS-stage C->A transpose,
// O += P@V (4 WMMAs). Epilogue multiplies each row by scale[h]/sqrt(t+1) and
// writes O into [B,T,H*HD] bf16 for the output GEMM.
// ---------------------------------------------------------------------------
__global__ void k_attn(const bf16_t* __restrict__ qn,
                       const bf16_t* __restrict__ kn,
                       const bf16_t* __restrict__ vT,
                       const float* __restrict__ scale,
                       bf16_t* __restrict__ obuf) {
  __shared__ float sst[4][16][32];   // per-wave S staging (f32)

  const int lane = threadIdx.x & 31;
  const int wv   = threadIdx.x >> 5;
  const int m    = lane & 15;
  const int hi   = (lane >> 4) & 1;

  const int bh = blockIdx.x;          // b*H + h
  const int h  = bh % Hc;
  const int bi = bh / Hc;
  const int t0 = blockIdx.y * 64 + wv * 16;

  const bf16_t* qbase = qn + (long)bh * Tc * NWc;
  const bf16_t* kbase = kn + (long)bh * Tc * NWc;
  const bf16_t* vbase = vT + (long)bh * HDc * Tc;

  // Q A-fragments (w = 0..31 and 32..63), resident for the whole loop.
  const v16bf a0 = afrag_bf16(qbase + (long)(t0 + m) * NWc + 0,  lane);
  const v16bf a1 = afrag_bf16(qbase + (long)(t0 + m) * NWc + 32, lane);

  v8f o[4];
#pragma unroll
  for (int dt = 0; dt < 4; ++dt) o[dt] = v8f_zero();

  const int tmax = t0 + 15;
  for (int s0 = 0; s0 <= tmax; s0 += 32) {
    // S tiles: columns s0..s0+15 and s0+16..s0+31
    v8f S[2];
#pragma unroll
    for (int st = 0; st < 2; ++st) {
      const bf16_t* kr = kbase + (long)(s0 + st * 16 + m) * NWc;
      v8f c = v8f_zero();
      c = wmma_bf16(a0, bfrag_bf16(kr + hi * 16),      c);
      c = wmma_bf16(a1, bfrag_bf16(kr + 32 + hi * 16), c);
      S[st] = c;
    }
    // Causal mask + stage into LDS as [row t_local][col s_local] f32.
#pragma unroll
    for (int st = 0; st < 2; ++st) {
      const int sg = s0 + st * 16 + m;
#pragma unroll
      for (int i = 0; i < 8; ++i) {
        const int tl = crow(lane, i);
        const float v = (sg <= t0 + tl) ? S[st][i] : 0.f;
        sst[wv][tl][st * 16 + m] = v;
      }
    }
    asm volatile("s_wait_dscnt 0" ::: "memory");   // wave-private staging: no barrier needed

    // P as A-fragment (16 rows x 32 s), converted to bf16.
    const v16bf pa = afrag_f32(&sst[wv][m][0], lane);

    // O += P @ V  over the four 16-wide d tiles.
#pragma unroll
    for (int dt = 0; dt < 4; ++dt) {
      const bf16_t* vr = vbase + (long)(dt * 16 + m) * Tc + s0 + hi * 16;
      o[dt] = wmma_bf16(pa, bfrag_bf16(vr), o[dt]);
    }
  }

  // Epilogue: row factor scale[h]/sqrt(t+1); write [B,T,H*HD] bf16.
  const float sc = scale[h];
#pragma unroll
  for (int i = 0; i < 8; ++i) {
    const int tg = t0 + crow(lane, i);
    const float fct = sc * rsqrtf((float)(tg + 1));
#pragma unroll
    for (int dt = 0; dt < 4; ++dt) {
      const int dcol = h * HDc + dt * 16 + m;
      obuf[((long)bi * Tc + tg) * Dc + dcol] = (bf16_t)(o[dt][i] * fct);
    }
  }
}

// ---------------------------------------------------------------------------
// Kernel 4: output projection  out[4096x1024] f32 = Obuf bf16 @ WoT bf16 + bo
// ---------------------------------------------------------------------------
__global__ void k_outproj(const bf16_t* __restrict__ Obuf,
                          const bf16_t* __restrict__ WoT,
                          const float* __restrict__ bo,
                          float* __restrict__ out) {
  const int lane = threadIdx.x & 31;
  const int wv   = threadIdx.x >> 5;
  const int m    = lane & 15;
  const int hi   = (lane >> 4) & 1;

  const long r0 = (long)blockIdx.x * 64 + (long)wv * 16;
  const int  n0 = blockIdx.y * 64;

  v8f c[4];
#pragma unroll
  for (int nt = 0; nt < 4; ++nt) c[nt] = v8f_zero();

  for (int k0 = 0; k0 < Dc; k0 += 32) {
    v16bf a = afrag_bf16(Obuf + (r0 + m) * Dc + k0, lane);
#pragma unroll
    for (int nt = 0; nt < 4; ++nt) {
      const int n = n0 + nt * 16 + m;
      c[nt] = wmma_bf16(a, bfrag_bf16(WoT + (long)n * Dc + k0 + hi * 16), c[nt]);
    }
  }

#pragma unroll
  for (int nt = 0; nt < 4; ++nt) {
    const int n = n0 + nt * 16 + m;
    const float bias = bo[n];
#pragma unroll
    for (int i = 0; i < 8; ++i) {
      out[(r0 + crow(lane, i)) * Dc + n] = c[nt][i] + bias;
    }
  }
}

// ---------------------------------------------------------------------------
// Launcher
// ---------------------------------------------------------------------------
extern "C" void kernel_launch(void* const* d_in, const int* in_sizes, int n_in,
                              void* d_out, int out_size, void* d_ws, size_t ws_size,
                              hipStream_t stream) {
  (void)in_sizes; (void)n_in; (void)out_size; (void)ws_size;

  const float* x     = (const float*)d_in[0];
  const float* Wqf   = (const float*)d_in[1];
  const float* bqf   = (const float*)d_in[2];
  const float* Wkf   = (const float*)d_in[3];
  const float* bkf   = (const float*)d_in[4];
  const float* Wqp   = (const float*)d_in[5];
  const float* bqp   = (const float*)d_in[6];
  const float* Wkp   = (const float*)d_in[7];
  const float* bkp   = (const float*)d_in[8];
  const float* Wv    = (const float*)d_in[9];
  const float* bv    = (const float*)d_in[10];
  const float* Wo    = (const float*)d_in[11];
  const float* bo    = (const float*)d_in[12];
  const float* scale = (const float*)d_in[13];
  float* out = (float*)d_out;

  // Workspace carve (all 256B aligned)
  char*  ws  = (char*)d_ws;
  size_t off = 0;
  auto carve = [&](size_t bytes) -> char* {
    char* p = ws + off;
    off += (bytes + 255) & ~(size_t)255;
    return p;
  };
  const size_t wbytes = (size_t)Dc * Dc * sizeof(bf16_t);          // 2 MB
  bf16_t* WqfT = (bf16_t*)carve(wbytes);
  bf16_t* WqpT = (bf16_t*)carve(wbytes);
  bf16_t* WkfT = (bf16_t*)carve(wbytes);
  bf16_t* WkpT = (bf16_t*)carve(wbytes);
  bf16_t* WvT  = (bf16_t*)carve(wbytes);
  bf16_t* WoT  = (bf16_t*)carve(wbytes);
  const size_t actbytes = (size_t)Bc * Hc * Tc * NWc * sizeof(bf16_t);  // 8 MB
  bf16_t* qnb  = (bf16_t*)carve(actbytes);
  bf16_t* knb  = (bf16_t*)carve(actbytes);
  bf16_t* vTb  = (bf16_t*)carve((size_t)Bc * Hc * HDc * Tc * sizeof(bf16_t));
  bf16_t* Obuf = (bf16_t*)carve((size_t)Bc * Tc * Dc * sizeof(bf16_t));

  // 0) weight convert+transpose (6x)
  {
    dim3 grid((Dc * Dc + 255) / 256), block(256);
    k_wcvt<<<grid, block, 0, stream>>>(Wqf, WqfT);
    k_wcvt<<<grid, block, 0, stream>>>(Wqp, WqpT);
    k_wcvt<<<grid, block, 0, stream>>>(Wkf, WkfT);
    k_wcvt<<<grid, block, 0, stream>>>(Wkp, WkpT);
    k_wcvt<<<grid, block, 0, stream>>>(Wv,  WvT);
    k_wcvt<<<grid, block, 0, stream>>>(Wo,  WoT);
  }

  // 1) wave projections (q then k)
  {
    dim3 grid(Bc * Tc / 64, Hc), block(128);
    k_waveproj<<<grid, block, 0, stream>>>(x, WqfT, WqpT, bqf, bqp, qnb);
    k_waveproj<<<grid, block, 0, stream>>>(x, WkfT, WkpT, bkf, bkp, knb);
  }

  // 2) v projection (transposed output)
  {
    dim3 grid(Bc * Tc / 64, Hc), block(128);
    k_vproj<<<grid, block, 0, stream>>>(x, WvT, bv, vTb);
  }

  // 3) causal wave attention
  {
    dim3 grid(Bc * Hc, Tc / 64), block(128);
    k_attn<<<grid, block, 0, stream>>>(qnb, knb, vTb, scale, Obuf);
  }

  // 4) output projection
  {
    dim3 grid(Bc * Tc / 64, Dc / 64), block(128);
    k_outproj<<<grid, block, 0, stream>>>(Obuf, WoT, bo, out);
  }
}
